// GSAVideoBlock_60129542155
// MI455X (gfx1250) — compile-verified, matrix operand order checked by reading, not run on hardware
//
#include <hip/hip_runtime.h>

#define BD 4
#define TD 512
#define DD 1024
#define HH 16
#define MLPD 4096

typedef __attribute__((ext_vector_type(16))) _Float16 v16h;
typedef __attribute__((ext_vector_type(8)))  _Float16 v8h;
typedef __attribute__((ext_vector_type(8)))  float    v8f;

// ---------------------------------------------------------------------------
// Transpose + f32->f16 convert:  in[K][N] (row-major) -> out[N][K] (row-major)
// so GEMM B operand is K-contiguous like A.
// ---------------------------------------------------------------------------
__global__ __launch_bounds__(256) void transpose_cvt_f16(
    const float* __restrict__ in, _Float16* __restrict__ out, int K, int N)
{
    __shared__ float tile[32][33];
    const int tx = threadIdx.x, ty = threadIdx.y;
    const int n0 = blockIdx.x * 32, k0 = blockIdx.y * 32;
    for (int j = 0; j < 32; j += 8)
        tile[ty + j][tx] = in[(size_t)(k0 + ty + j) * N + (n0 + tx)];
    __syncthreads();
    for (int j = 0; j < 32; j += 8)
        out[(size_t)(n0 + ty + j) * K + (k0 + tx)] = (_Float16)tile[tx][ty + j];
}

// ---------------------------------------------------------------------------
// LayerNorm over D=1024, output f16 (feeds WMMA GEMM A operand)
// ---------------------------------------------------------------------------
__global__ __launch_bounds__(256) void layernorm_f16(
    const float* __restrict__ x, const float* __restrict__ w,
    const float* __restrict__ b, _Float16* __restrict__ out)
{
    __shared__ float s1[256], s2[256];
    const int row = blockIdx.x, tid = threadIdx.x;
    const float* xr = x + (size_t)row * DD;
    float xs[4], sum = 0.f, sq = 0.f;
    for (int j = 0; j < 4; ++j) {
        float v = xr[tid + 256 * j];
        xs[j] = v; sum += v; sq += v * v;
    }
    s1[tid] = sum; s2[tid] = sq;
    __syncthreads();
    for (int st = 128; st > 0; st >>= 1) {
        if (tid < st) { s1[tid] += s1[tid + st]; s2[tid] += s2[tid + st]; }
        __syncthreads();
    }
    const float mu   = s1[0] * (1.f / DD);
    const float var  = s2[0] * (1.f / DD) - mu * mu;
    const float rstd = rsqrtf(var + 1e-6f);
    _Float16* orow = out + (size_t)row * DD;
    for (int j = 0; j < 4; ++j) {
        int i = tid + 256 * j;
        orow[i] = (_Float16)((xs[j] - mu) * rstd * w[i] + b[i]);
    }
}

// ---------------------------------------------------------------------------
// WMMA GEMM: C[M x N] = epilogue(A[M x K] @ Bt[N x K]^T)
// 128x128 block tile, 256 threads = 8 waves, each wave 32x64 (2x4 of 16x16).
// K staged 32-deep, DOUBLE-BUFFERED in LDS via CDNA5 async global->LDS copies
// (global_load_async_to_lds_b128, ASYNCcnt), so the HBM fetch of stage k+1
// overlaps the v_wmma work on stage k.
// EPI: 0=store f32, 1=swish f32, 2=gate (s,g) f32, 3=+res f32,
//      4=+bias,gelu -> f16, 5=+bias,+res f32
// ---------------------------------------------------------------------------
template<int EPI>
__global__ __launch_bounds__(256) void gemm_wmma(
    const _Float16* __restrict__ A, const _Float16* __restrict__ Bt,
    float* __restrict__ out0, float* __restrict__ out1,
    _Float16* __restrict__ outh,
    const float* __restrict__ bias, const float* __restrict__ res,
    int N, int K)
{
    __shared__ _Float16 As[2][128 * 32];
    __shared__ _Float16 Bs[2][128 * 32];
    const int tid  = threadIdx.x;
    const int lane = tid & 31, wid = tid >> 5;
    const int wm = wid >> 1, wn = wid & 1;       // 4 waves along M, 2 along N
    const int lr = lane & 15, hs = lane >> 4;    // half-wave select (ISA layout)
    const int rowBlk = blockIdx.y * 128, colBlk = blockIdx.x * 128;

    // Each thread owns two fixed 16-byte chunks of the A tile and of the B
    // tile per stage: chunk c in [0,512): row = c>>2, half-offset = (c&3)*8.
    const int c0 = tid, c1 = tid + 256;
    const int r0 = c0 >> 2, o0 = (c0 & 3) * 8;
    const int r1 = c1 >> 2, o1 = (c1 & 3) * 8;
    const _Float16* gA0 = A  + (size_t)(rowBlk + r0) * K + o0;
    const _Float16* gA1 = A  + (size_t)(rowBlk + r1) * K + o1;
    const _Float16* gB0 = Bt + (size_t)(colBlk + r0) * K + o0;
    const _Float16* gB1 = Bt + (size_t)(colBlk + r1) * K + o1;
    // LDS byte addresses (low 32 bits of the generic pointer = LDS offset)
    const unsigned lA0 = (unsigned)(uintptr_t)&As[0][r0 * 32 + o0];
    const unsigned lA1 = (unsigned)(uintptr_t)&As[0][r1 * 32 + o1];
    const unsigned lB0 = (unsigned)(uintptr_t)&Bs[0][r0 * 32 + o0];
    const unsigned lB1 = (unsigned)(uintptr_t)&Bs[0][r1 * 32 + o1];
    const unsigned bufBytes = 128 * 32 * 2;

    auto issue_stage = [&](int kk, int buf) {
        const unsigned bo = (unsigned)buf * bufBytes;
        const size_t ko = (size_t)kk * 32;
        asm volatile(
            "global_load_async_to_lds_b128 %0, %4, off\n\t"
            "global_load_async_to_lds_b128 %1, %5, off\n\t"
            "global_load_async_to_lds_b128 %2, %6, off\n\t"
            "global_load_async_to_lds_b128 %3, %7, off"
            :
            : "v"(lA0 + bo), "v"(lA1 + bo), "v"(lB0 + bo), "v"(lB1 + bo),
              "v"((unsigned long long)(uintptr_t)(gA0 + ko)),
              "v"((unsigned long long)(uintptr_t)(gA1 + ko)),
              "v"((unsigned long long)(uintptr_t)(gB0 + ko)),
              "v"((unsigned long long)(uintptr_t)(gB1 + ko))
            : "memory");
    };

    v8f acc[2][4];
    for (int i = 0; i < 2; ++i)
        for (int j = 0; j < 4; ++j)
            for (int r = 0; r < 8; ++r) acc[i][j][r] = 0.f;

    const int nk = K >> 5;
    issue_stage(0, 0);
    for (int kk = 0; kk < nk; ++kk) {
        const int buf = kk & 1;
        if (kk + 1 < nk) {
            issue_stage(kk + 1, buf ^ 1);                       // prefetch next
            asm volatile("s_wait_asynccnt 0x4" ::: "memory");   // stage kk done
        } else {
            asm volatile("s_wait_asynccnt 0x0" ::: "memory");
        }
        __syncthreads();

        const _Float16* Ab = As[buf];
        const _Float16* Bb = Bs[buf];
        // Fragment layout (16-bit A 16x32): lanes 0-15 hold K 0-7 / 16-23,
        // lanes 16-31 hold K 8-15 / 24-31, for row M = lane%16.
        v16h af[2], bf[4];
        for (int i = 0; i < 2; ++i) {
            int r = wm * 32 + i * 16 + lr;
            v8h lo = *(const v8h*)&Ab[r * 32 + hs * 8];
            v8h hi = *(const v8h*)&Ab[r * 32 + hs * 8 + 16];
            af[i] = __builtin_shufflevector(lo, hi, 0,1,2,3,4,5,6,7,8,9,10,11,12,13,14,15);
        }
        for (int j = 0; j < 4; ++j) {
            int r = wn * 64 + j * 16 + lr;
            v8h lo = *(const v8h*)&Bb[r * 32 + hs * 8];
            v8h hi = *(const v8h*)&Bb[r * 32 + hs * 8 + 16];
            bf[j] = __builtin_shufflevector(lo, hi, 0,1,2,3,4,5,6,7,8,9,10,11,12,13,14,15);
        }
        for (int i = 0; i < 2; ++i)
            for (int j = 0; j < 4; ++j)
                acc[i][j] = __builtin_amdgcn_wmma_f32_16x16x32_f16(
                    false, af[i], false, bf[j], (short)0, acc[i][j], false, false);
        __syncthreads();   // all LDS reads of buf done -> safe to refill later
    }

    // Epilogue. C/D layout: VGPR r holds M = r (lanes 0-15) or r+8 (lanes 16-31),
    // N = lane%16.
    for (int i = 0; i < 2; ++i)
        for (int j = 0; j < 4; ++j)
            for (int r = 0; r < 8; ++r) {
                int row = rowBlk + wm * 32 + i * 16 + hs * 8 + r;
                int col = colBlk + wn * 64 + j * 16 + lr;
                size_t idx = (size_t)row * N + col;
                float v = acc[i][j][r];
                if (EPI == 0) {
                    out0[idx] = v;
                } else if (EPI == 1) {                     // swish
                    out0[idx] = v / (1.f + __expf(-v));
                } else if (EPI == 2) {                     // gate: g = logsig(v)/8, s = 1-exp(g)
                    float ls = fminf(v, 0.f) - log1pf(__expf(-fabsf(v)));
                    float gg = ls * 0.125f;
                    out1[idx] = gg;
                    out0[idx] = -expm1f(gg);
                } else if (EPI == 3) {                     // + residual
                    out0[idx] = v + res[idx];
                } else if (EPI == 4) {                     // + bias, exact gelu, f16 out
                    float t = v + bias[col];
                    outh[idx] = (_Float16)(0.5f * t * (1.f + erff(t * 0.70710678118f)));
                } else if (EPI == 5) {                     // + bias + residual
                    out0[idx] = v + bias[col] + res[idx];
                }
            }
}

// ---------------------------------------------------------------------------
// Gated-slot attention scan. One block per (b,h); K[64x64] and V[64x64]
// state resident in LDS. 256 threads; reductions 4-way parallel over the
// contraction dim with LDS partial sums to shorten the serial chain.
// ---------------------------------------------------------------------------
__global__ __launch_bounds__(256) void gsa_scan_kernel(
    const float* __restrict__ q, const float* __restrict__ k,
    const float* __restrict__ v, const float* __restrict__ s,
    const float* __restrict__ g, float* __restrict__ o)
{
    __shared__ float Ks[64][65];    // [d][m], padded
    __shared__ float Vs[64][65];    // [m][d], padded
    __shared__ float sq[64], sk[64], sv[64], ss[64], sdec[64];
    __shared__ float sok[64], sqv[64];
    __shared__ float spart[256];

    const int tid = threadIdx.x;
    const int b = blockIdx.x / HH, h = blockIdx.x % HH;
    for (int i = tid; i < 64 * 65; i += 256) {
        (&Ks[0][0])[i] = 0.f;
        (&Vs[0][0])[i] = 0.f;
    }
    __syncthreads();

    const int mcol = tid & 63, seg = tid >> 6;      // reduction lane / segment
    const int rOwn = tid >> 2, c0 = (tid & 3) * 16; // state-update ownership

    for (int t = 0; t < TD; ++t) {
        const size_t off = ((size_t)(b * TD + t) * HH + h) * 64;
        if (tid < 64)        sq[tid]        = q[off + tid];
        else if (tid < 128)  sk[tid - 64]   = k[off + tid - 64];
        else if (tid < 192)  sv[tid - 128]  = v[off + tid - 128];
        else { int m = tid - 192; sdec[m] = __expf(g[off + m]); ss[m] = s[off + m]; }
        __syncthreads();

        { // K[d][m] = K*dec[m] + k[d]*s[m]
            float kd = sk[rOwn];
            for (int i = 0; i < 16; ++i) {
                int m = c0 + i;
                Ks[rOwn][m] = Ks[rOwn][m] * sdec[m] + kd * ss[m];
            }
        }
        __syncthreads();

        { // ok[m] = scale * sum_d q[d]*K[d][m]  (4 partial segments)
            float p = 0.f; int d0 = seg * 16;
            for (int i = 0; i < 16; ++i) p += sq[d0 + i] * Ks[d0 + i][mcol];
            spart[tid] = p;
        }
        __syncthreads();
        if (tid < 64)
            sok[tid] = 0.125f * (spart[tid] + spart[tid + 64] + spart[tid + 128] + spart[tid + 192]);
        __syncthreads();
        if (tid < 64) { // softmax over 64 slots
            float mx = -3.4e38f;
            for (int j = 0; j < 64; ++j) mx = fmaxf(mx, sok[j]);
            spart[tid] = __expf(sok[tid] - mx);
        }
        __syncthreads();
        if (tid < 64) {
            float sum = 0.f;
            for (int j = 0; j < 64; ++j) sum += spart[j];
            sqv[tid] = spart[tid] / sum;
        }
        __syncthreads();

        { // V[m][d] = V*dec[m] + s[m]*v[d]
            float dec = sdec[rOwn], sm = ss[rOwn];
            for (int i = 0; i < 16; ++i) {
                int d = c0 + i;
                Vs[rOwn][d] = Vs[rOwn][d] * dec + sm * sv[d];
            }
        }
        __syncthreads();

        { // o[d] = sum_m qv[m]*V[m][d]
            float p = 0.f; int m0 = seg * 16;
            for (int i = 0; i < 16; ++i) p += sqv[m0 + i] * Vs[m0 + i][mcol];
            spart[tid] = p;
        }
        __syncthreads();
        if (tid < 64)
            o[off + tid] = spart[tid] + spart[tid + 64] + spart[tid + 128] + spart[tid + 192];
        __syncthreads();
    }
}

// ---------------------------------------------------------------------------
// Per-(b,t,h) RMSNorm over DV=64, convert to f16 for the Wo GEMM.
// ---------------------------------------------------------------------------
__global__ __launch_bounds__(64) void rms_cvt_kernel(
    const float* __restrict__ o, const float* __restrict__ w,
    _Float16* __restrict__ out)
{
    __shared__ float red[64];
    const int row = blockIdx.x, tid = threadIdx.x;
    float x = o[(size_t)row * 64 + tid];
    red[tid] = x * x;
    __syncthreads();
    for (int st = 32; st > 0; st >>= 1) {
        if (tid < st) red[tid] += red[tid + st];
        __syncthreads();
    }
    float rstd = rsqrtf(red[0] * (1.f / 64.f) + 1e-6f);
    out[(size_t)row * 64 + tid] = (_Float16)(x * rstd * w[tid]);
}

// ---------------------------------------------------------------------------
extern "C" void kernel_launch(void* const* d_in, const int* in_sizes, int n_in,
                              void* d_out, int out_size, void* d_ws, size_t ws_size,
                              hipStream_t stream)
{
    const float* x    = (const float*)d_in[0];
    const float* ln1w = (const float*)d_in[1];
    const float* ln1b = (const float*)d_in[2];
    const float* Wq   = (const float*)d_in[3];
    const float* Wk   = (const float*)d_in[4];
    const float* Wv   = (const float*)d_in[5];
    const float* Wf   = (const float*)d_in[6];
    const float* gnw  = (const float*)d_in[7];
    const float* Wo   = (const float*)d_in[8];
    const float* ln2w = (const float*)d_in[9];
    const float* ln2b = (const float*)d_in[10];
    const float* W1   = (const float*)d_in[11];
    const float* b1   = (const float*)d_in[12];
    const float* W2   = (const float*)d_in[13];
    const float* b2   = (const float*)d_in[14];
    float* outp = (float*)d_out;

    const int BT = BD * TD;   // 2048 rows for every GEMM

    char* wsp = (char*)d_ws;
    auto carve = [&](size_t bytes) -> char* {
        char* p = wsp;
        wsp += (bytes + 255) & ~(size_t)255;
        return p;
    };
    _Float16* h16   = (_Float16*)carve((size_t)BT * DD * 2);
    _Float16* WqT   = (_Float16*)carve((size_t)DD * 1024 * 2);
    _Float16* WkT   = (_Float16*)carve((size_t)DD * 1024 * 2);
    _Float16* WvT   = (_Float16*)carve((size_t)DD * 1024 * 2);
    _Float16* WfT   = (_Float16*)carve((size_t)DD * 1024 * 2);
    _Float16* WoT   = (_Float16*)carve((size_t)DD * 1024 * 2);
    _Float16* W1T   = (_Float16*)carve((size_t)MLPD * DD * 2);
    _Float16* W2T   = (_Float16*)carve((size_t)DD * MLPD * 2);
    float*    qb    = (float*)carve((size_t)BT * 1024 * 4);
    float*    kb    = (float*)carve((size_t)BT * 1024 * 4);
    float*    vb    = (float*)carve((size_t)BT * 1024 * 4);
    float*    sb    = (float*)carve((size_t)BT * 1024 * 4);
    float*    gb    = (float*)carve((size_t)BT * 1024 * 4);
    float*    ob    = (float*)carve((size_t)BT * 1024 * 4);
    _Float16* on16  = (_Float16*)carve((size_t)BT * 1024 * 2);
    float*    yb    = (float*)carve((size_t)BT * DD * 4);
    _Float16* h216  = (_Float16*)carve((size_t)BT * DD * 2);
    _Float16* a16   = (_Float16*)carve((size_t)BT * MLPD * 2);

    const dim3 tb(32, 8);
    // weight transposes: in[K][N] -> f16 [N][K]
    transpose_cvt_f16<<<dim3(1024 / 32, DD / 32),  tb, 0, stream>>>(Wq, WqT, DD, 1024);
    transpose_cvt_f16<<<dim3(1024 / 32, DD / 32),  tb, 0, stream>>>(Wk, WkT, DD, 1024);
    transpose_cvt_f16<<<dim3(1024 / 32, DD / 32),  tb, 0, stream>>>(Wv, WvT, DD, 1024);
    transpose_cvt_f16<<<dim3(1024 / 32, DD / 32),  tb, 0, stream>>>(Wf, WfT, DD, 1024);
    transpose_cvt_f16<<<dim3(DD / 32,   1024 / 32),tb, 0, stream>>>(Wo, WoT, 1024, DD);
    transpose_cvt_f16<<<dim3(MLPD / 32, DD / 32),  tb, 0, stream>>>(W1, W1T, DD, MLPD);
    transpose_cvt_f16<<<dim3(DD / 32,   MLPD / 32),tb, 0, stream>>>(W2, W2T, MLPD, DD);

    // LN1
    layernorm_f16<<<BT, 256, 0, stream>>>(x, ln1w, ln1b, h16);

    // projections (fused activations)
    const dim3 gD(1024 / 128, BT / 128);
    gemm_wmma<1><<<gD, 256, 0, stream>>>(h16, WqT, qb, nullptr, nullptr, nullptr, nullptr, 1024, DD);
    gemm_wmma<1><<<gD, 256, 0, stream>>>(h16, WkT, kb, nullptr, nullptr, nullptr, nullptr, 1024, DD);
    gemm_wmma<0><<<gD, 256, 0, stream>>>(h16, WvT, vb, nullptr, nullptr, nullptr, nullptr, 1024, DD);
    gemm_wmma<2><<<gD, 256, 0, stream>>>(h16, WfT, sb, gb,      nullptr, nullptr, nullptr, 1024, DD);

    // recurrent gated-slot scan
    gsa_scan_kernel<<<BD * HH, 256, 0, stream>>>(qb, kb, vb, sb, gb, ob);

    // per-head RMSNorm -> f16
    rms_cvt_kernel<<<BT * HH, 64, 0, stream>>>(ob, gnw, on16);

    // output projection + residual
    gemm_wmma<3><<<gD, 256, 0, stream>>>(on16, WoT, yb, nullptr, nullptr, nullptr, x, DD, 1024);

    // LN2
    layernorm_f16<<<BT, 256, 0, stream>>>(yb, ln2w, ln2b, h216);

    // MLP
    gemm_wmma<4><<<dim3(MLPD / 128, BT / 128), 256, 0, stream>>>(
        h216, W1T, nullptr, nullptr, a16, b1, nullptr, MLPD, DD);
    gemm_wmma<5><<<dim3(DD / 128, BT / 128), 256, 0, stream>>>(
        a16, W2T, outp, nullptr, nullptr, b2, yb, DD, MLPD);
}
